// NoTime_20547123544828
// MI455X (gfx1250) — matrix operand order repaired and checked by reading, not compiled, not run on hardware
//
#include <hip/hip_runtime.h>
#include <hip/hip_fp16.h>
#include <hip/hip_bf16.h>

// ---------------------------------------------------------------------------
// CDNA5 (gfx1250) implementation. All matmuls >= 128x256x256 go through
// v_wmma_f32_16x16x32_f16 (f16 inputs, f32 accumulate). Weights are packed
// once per launch into [N,K] f16 so B fragments are two 16B loads from one Wt row.
// ---------------------------------------------------------------------------

typedef _Float16 half_t;
typedef __attribute__((ext_vector_type(16))) _Float16 v16h;
typedef __attribute__((ext_vector_type(8)))  _Float16 v8h;
typedef __attribute__((ext_vector_type(8)))  float    v8f;

static __device__ __forceinline__ float sigmoidf_(float x) {
    return 1.f / (1.f + __expf(-x));
}

// Block-wide layernorm over 256 elements (one row per 256-thread block).
static __device__ __forceinline__ float block_ln256(float v, const float* g,
                                                    const float* be, float* sbuf,
                                                    int t) {
    sbuf[t] = v; __syncthreads();
    for (int off = 128; off > 0; off >>= 1) {
        if (t < off) sbuf[t] += sbuf[t + off];
        __syncthreads();
    }
    float mean = sbuf[0] * (1.f / 256.f);
    __syncthreads();
    float d = v - mean;
    sbuf[t] = d * d; __syncthreads();
    for (int off = 128; off > 0; off >>= 1) {
        if (t < off) sbuf[t] += sbuf[t + off];
        __syncthreads();
    }
    float var = sbuf[0] * (1.f / 256.f);
    __syncthreads();
    return d * rsqrtf(var + 1e-5f) * g[t] + be[t];
}

// ---------------------------------------------------------------------------
// Generic WMMA GEMM: C[M,N] = act(A[M,K](f16) @ Wt[N,K]^T(f16) + bias)
// One wave per 16x16 tile; block = 4 waves covering 64 columns.
// A fragment (16-bit 16x32): lane m=l&15, kb=(l&16?8:0); halfs 0..7 = K kb..kb+7,
// halfs 8..15 = K kb+16..kb+23.  B fragment (32x16): lane n=l&15,
// kb=(l&16?8:0); halfs 0..7 = K kb..kb+7, halfs 8..15 = K kb+16..kb+23.
// ---------------------------------------------------------------------------
__global__ void gemm_f16_wmma(const half_t* __restrict__ A,
                              const half_t* __restrict__ Wt,
                              const float* __restrict__ bias,
                              float* __restrict__ C, half_t* __restrict__ C16,
                              int M, int K, int N, int act) {
    const int lane = threadIdx.x & 31;
    const int wave = threadIdx.x >> 5;
    const int n0 = (blockIdx.x * 4 + wave) * 16;
    const int m0 = blockIdx.y * 16;
    if (n0 >= N) return;  // wave-uniform: EXEC stays all-ones for WMMA

    const int am  = m0 + (lane & 15);
    const int akb = (lane & 16) ? 8 : 0;
    const int bn  = n0 + (lane & 15);
    const int bkb = (lane & 16) ? 8 : 0;
    const half_t* arow = A  + (size_t)am * K;
    const half_t* brow = Wt + (size_t)bn * K;

    v8f acc = {0.f, 0.f, 0.f, 0.f, 0.f, 0.f, 0.f, 0.f};
    for (int k0 = 0; k0 < K; k0 += 32) {
        const v8h* pa = (const v8h*)(arow + k0 + akb);
        v8h a0 = pa[0];  // K = k0+akb .. +7
        v8h a1 = pa[2];  // K = k0+akb+16 .. +23
        v16h a;
#pragma unroll
        for (int j = 0; j < 8; ++j) { a[j] = a0[j]; a[8 + j] = a1[j]; }
        const v8h* pb = (const v8h*)(brow + k0 + bkb);
        v8h b0 = pb[0], b1 = pb[2];
        v16h b;
#pragma unroll
        for (int j = 0; j < 8; ++j) { b[j] = b0[j]; b[8 + j] = b1[j]; }
        acc = __builtin_amdgcn_wmma_f32_16x16x32_f16(
            false, a, false, b, (short)0, acc, false, false);
    }

    const int col   = n0 + (lane & 15);
    const int rbase = m0 + ((lane & 16) ? 8 : 0);
    const float bv  = bias ? bias[col] : 0.f;
#pragma unroll
    for (int r = 0; r < 8; ++r) {
        int row = rbase + r;
        float v = acc[r] + bv;
        if (act == 1) v = fmaxf(v, 0.f);
        else if (act == 2) v = tanhf(v);
        if (row < M) {
            C[(size_t)row * N + col] = v;
            if (C16) C16[(size_t)row * N + col] = (half_t)v;
        }
    }
}

// W[K,N] f32 -> Wt[N,K] f16
__global__ void transpose_f16(const float* __restrict__ W,
                              half_t* __restrict__ Wt, int K, int N) {
    long long total = (long long)K * N;
    for (long long idx = (long long)blockIdx.x * blockDim.x + threadIdx.x;
         idx < total; idx += (long long)gridDim.x * blockDim.x) {
        int k = (int)(idx / N), n = (int)(idx % N);
        Wt[(size_t)n * K + k] = (half_t)W[idx];
    }
}

// Build concatenated f16 activation matrix A[M,K] from up to 4 f32 sources
// (rowstride 0 => broadcast one row). Rows >= Mvalid are zero-padded.
__global__ void pack_concat(half_t* __restrict__ A, long long total, int K,
                            int Mvalid,
                            const float* s0, int w0, int r0,
                            const float* s1, int w1, int r1,
                            const float* s2, int w2, int r2,
                            const float* s3, int w3, int r3) {
    for (long long idx = (long long)blockIdx.x * blockDim.x + threadIdx.x;
         idx < total; idx += (long long)gridDim.x * blockDim.x) {
        int m = (int)(idx / K), k = (int)(idx % K);
        float v = 0.f;
        if (m < Mvalid) {
            if (k < w0) v = s0[(long long)m * r0 + k];
            else {
                k -= w0;
                if (k < w1) v = s1[(long long)m * r1 + k];
                else {
                    k -= w1;
                    if (k < w2) v = s2[(long long)m * r2 + k];
                    else {
                        k -= w2;
                        if (s3 && k < w3) v = s3[(long long)m * r3 + k];
                    }
                }
            }
        }
        A[idx] = (half_t)v;
    }
}

// Code-selection gate: logits = T @ cs2 + b ; pmask = 1 if argmax==0
__global__ void code_gate2(const float* __restrict__ T,
                           const float* __restrict__ w2,
                           const float* __restrict__ b2,
                           float* __restrict__ pmask, int Mvalid) {
    int v = blockIdx.x * blockDim.x + threadIdx.x;
    if (v >= Mvalid) return;
    const float* row = T + (size_t)v * 256;
    float l0 = b2[0], l1 = b2[1];
    for (int d = 0; d < 256; ++d) {
        float t = row[d];
        l0 += t * w2[d * 2 + 0];
        l1 += t * w2[d * 2 + 1];
    }
    pmask[v] = (l0 >= l1) ? 1.f : 0.f;
}

// selected[b,s,:] = sum_c pmask[idx] * relu(emb[idx])
__global__ void build_selected(const int* __restrict__ seqs,
                               const float* __restrict__ emb,
                               const float* __restrict__ pmask,
                               float* __restrict__ selected) {
    __shared__ int   ids[32];
    __shared__ float pm[32];
    int bs = blockIdx.x, t = threadIdx.x;
    if (t < 32) {
        int id = seqs[bs * 32 + t];
        ids[t] = id;
        pm[t]  = pmask[id];
    }
    __syncthreads();
    float acc = 0.f;
    for (int c = 0; c < 32; ++c) {
        float p = pm[c];
        if (p != 0.f) acc += p * fmaxf(emb[(size_t)ids[c] * 256 + t], 0.f);
    }
    selected[(size_t)bs * 256 + t] = acc;
}

// Reverse LSTM, bias-free, gates i,f,g,o; H=32. One block per batch row.
__global__ void lstm_rev(const float* __restrict__ selected,
                         const float* __restrict__ wih,   // [256,128]
                         const float* __restrict__ whh,   // [32,128]
                         float* __restrict__ following) { // [B,S,32]
    __shared__ float h[32], c[32], xs[256], gates[128];
    int t = threadIdx.x, b = blockIdx.x;
    if (t < 32) { h[t] = 0.f; c[t] = 0.f; }
    __syncthreads();
    for (int s = 63; s >= 0; --s) {
        const float* xr = selected + ((size_t)b * 64 + s) * 256;
        xs[t] = xr[t]; xs[t + 128] = xr[t + 128];
        __syncthreads();
        float g = 0.f;
        for (int k = 0; k < 256; ++k) g += xs[k] * wih[k * 128 + t];
        for (int j = 0; j < 32; ++j)  g += h[j] * whh[j * 128 + t];
        gates[t] = g;
        __syncthreads();
        if (t < 32) {
            float i_ = sigmoidf_(gates[t]);
            float f_ = sigmoidf_(gates[32 + t]);
            float g_ = tanhf(gates[64 + t]);
            float o_ = sigmoidf_(gates[96 + t]);
            float cn = f_ * c[t] + i_ * g_;
            c[t] = cn;
            h[t] = o_ * tanhf(cn);
            following[((size_t)b * 64 + s) * 32 + t] = h[t];
        }
        __syncthreads();
    }
}

// tgt1 = tl(tgt), tgt2 = tl2(tgt)
__global__ void tl_kernel(const float* __restrict__ tgt,
                          const float* __restrict__ tlw, const float* __restrict__ tlb,
                          const float* __restrict__ tl2w, const float* __restrict__ tl2b,
                          float* __restrict__ tgt1, float* __restrict__ tgt2) {
    int d = threadIdx.x;
    float a = tlb[d], b2 = tl2b[d];
    for (int k = 0; k < 256; ++k) {
        float tv = tgt[k];
        a  += tv * tlw[k * 256 + d];
        b2 += tv * tl2w[k * 256 + d];
    }
    tgt1[d] = a;
    tgt2[d] = b2;
}

// skip[b] = 1 iff argmax(v)==1 (keep hx)
__global__ void vskip_kernel(const float* __restrict__ T,
                             const float* __restrict__ w2,
                             const float* __restrict__ b2,
                             float* __restrict__ skip) {
    int b = threadIdx.x;
    const float* row = T + (size_t)b * 256;
    float v0 = b2[0], v1 = b2[1];
    for (int d = 0; d < 256; ++d) {
        float t = row[d];
        v0 += t * w2[d * 2 + 0];
        v1 += t * w2[d * 2 + 1];
    }
    skip[b] = (v1 > v0) ? 1.f : 0.f;
}

// Attention over C codes + gather attended vector (input to fc GEMM).
__global__ void attn_fc_in(const int* __restrict__ seqs,
                           const float* __restrict__ emb,
                           const float* __restrict__ pmaskV,
                           const float* __restrict__ q, int i,
                           half_t* __restrict__ Az) {
    __shared__ float qv[256];
    __shared__ float e[32];
    __shared__ float red[256];
    __shared__ int   ids[32];
    int b = blockIdx.x, t = threadIdx.x;
    qv[t] = q[(size_t)b * 256 + t];
    if (t < 32) ids[t] = seqs[((size_t)b * 64 + i) * 32 + t];
    __syncthreads();
    int c = t >> 3, l8 = t & 7;
    int id = ids[c];
    float part = 0.f;
    for (int d = l8; d < 256; d += 8)
        part += qv[d] * fmaxf(emb[(size_t)id * 256 + d], 0.f);
    red[t] = part; __syncthreads();
    if (l8 < 4) red[t] += red[t + 4];
    __syncthreads();
    if (l8 < 2) red[t] += red[t + 2];
    __syncthreads();
    if (l8 == 0) e[c] = (red[t] + red[t + 1]) * 0.0625f;  // / sqrt(256)
    __syncthreads();
    if (t == 0) {
        float mx = -3.4e38f;
        for (int j = 0; j < 32; ++j) mx = fmaxf(mx, e[j]);
        float sum = 0.f;
        for (int j = 0; j < 32; ++j) { float ex = __expf(e[j] - mx); e[j] = ex; sum += ex; }
        float inv = 1.f / sum;
        for (int j = 0; j < 32; ++j) e[j] = e[j] * inv * pmaskV[ids[j]];
    }
    __syncthreads();
    float acc = 0.f;
    for (int c2 = 0; c2 < 32; ++c2) {
        float a = e[c2];
        if (a != 0.f) acc += a * fmaxf(emb[(size_t)ids[c2] * 256 + t], 0.f);
    }
    Az[(size_t)b * 256 + t] = (half_t)acc;
}

// GRU combine (r,z,n gates) + straight-through skip mix
__global__ void gru_combine(const float* __restrict__ GI,
                            const float* __restrict__ GH,
                            const float* __restrict__ hx,
                            const float* __restrict__ skip,
                            float* __restrict__ shx, half_t* __restrict__ shxh) {
    int idx = blockIdx.x * blockDim.x + threadIdx.x;
    int b = idx >> 8, d = idx & 255;
    const float* gi = GI + (size_t)b * 768;
    const float* gh = GH + (size_t)b * 768;
    float r  = sigmoidf_(gi[d] + gh[d]);
    float zt = sigmoidf_(gi[256 + d] + gh[256 + d]);
    float n  = tanhf(gi[512 + d] + r * gh[512 + d]);
    float h  = hx[idx];
    float gru = (1.f - zt) * n + zt * h;
    float s = skip[b];
    float out = h * s + gru * (1.f - s);
    shx[idx]  = out;
    shxh[idx] = (half_t)out;
}

// Row layernorm (optional residual, strided f32 out, optional f16 out)
__global__ void layernorm256(const float* __restrict__ X,
                             const float* __restrict__ R,
                             const float* __restrict__ g,
                             const float* __restrict__ be,
                             float* __restrict__ Y, half_t* __restrict__ Yh,
                             long long ostride) {
    __shared__ float red[256];
    int m = blockIdx.x, t = threadIdx.x;
    float v = X[(size_t)m * 256 + t];
    if (R) v += R[(size_t)m * 256 + t];
    float y = block_ln256(v, g, be, red, t);
    Y[(size_t)m * ostride + t] = y;
    if (Yh) Yh[(size_t)m * 256 + t] = (half_t)y;
}

// Attention over growing memory + ln2
__global__ void mem_attention(const float* __restrict__ sh,
                              const float* __restrict__ memory,
                              const float* __restrict__ g2,
                              const float* __restrict__ b2, int i,
                              float* __restrict__ h2, half_t* __restrict__ h2h) {
    __shared__ float shl[256];
    __shared__ float me[64];
    __shared__ float red[256];
    int b = blockIdx.x, t = threadIdx.x;
    shl[t] = sh[(size_t)b * 256 + t];
    __syncthreads();
    int s = t >> 2, l4 = t & 3;
    bool valid = (s < i) || (s == 0);
    float part = 0.f;
    if (valid) {
        const float* mrow = memory + ((size_t)b * 64 + s) * 256;
        for (int d = l4; d < 256; d += 4) part += shl[d] * mrow[d];
    }
    red[t] = part; __syncthreads();
    if (l4 < 2) red[t] += red[t + 2];
    __syncthreads();
    if (l4 == 0) me[s] = valid ? (red[t] + red[t + 1]) * 0.0625f : -1e9f;
    __syncthreads();
    if (t == 0) {
        float mx = -3.4e38f;
        for (int j = 0; j < 64; ++j) mx = fmaxf(mx, me[j]);
        float sum = 0.f;
        for (int j = 0; j < 64; ++j) { float ex = __expf(me[j] - mx); me[j] = ex; sum += ex; }
        float inv = 1.f / sum;
        for (int j = 0; j < 64; ++j) me[j] *= inv;
    }
    __syncthreads();
    float acc = 0.f;
    for (int j = 0; j < 64; ++j)
        acc += me[j] * memory[((size_t)b * 64 + j) * 256 + t];
    float v = acc + shl[t];
    float y = block_ln256(v, g2, b2, red, t);
    h2[(size_t)b * 256 + t]  = y;
    h2h[(size_t)b * 256 + t] = (half_t)y;
}

// hx_new = (i==0) ? step_hx : LN(F2 + h2); write hx (f32+f16) and hiddens[:,i]
__global__ void step_final(const float* __restrict__ F2,
                           const float* __restrict__ h2,
                           const float* __restrict__ lng,
                           const float* __restrict__ lnb,
                           const float* __restrict__ shx, int i,
                           float* __restrict__ hx, half_t* __restrict__ hxh,
                           float* __restrict__ hiddens) {
    __shared__ float red[256];
    int b = blockIdx.x, t = threadIdx.x;
    float v = F2[(size_t)b * 256 + t] + h2[(size_t)b * 256 + t];
    float y = block_ln256(v, lng, lnb, red, t);
    float out = (i == 0) ? shx[(size_t)b * 256 + t] : y;
    hx[(size_t)b * 256 + t]  = out;
    hxh[(size_t)b * 256 + t] = (half_t)out;
    hiddens[((size_t)b * 64 + i) * 256 + t] = out;
}

// constant memory row = LN(relu(mem_b))
__global__ void memrow_kernel(const float* __restrict__ memb,
                              const float* __restrict__ g,
                              const float* __restrict__ be,
                              float* __restrict__ row) {
    __shared__ float red[256];
    int t = threadIdx.x;
    float v = fmaxf(memb[t], 0.f);
    row[t] = block_ln256(v, g, be, red, t);
}

__global__ void fill_broadcast(const float* __restrict__ row,
                               float* __restrict__ dst, long long total) {
    for (long long idx = (long long)blockIdx.x * blockDim.x + threadIdx.x;
         idx < total; idx += (long long)gridDim.x * blockDim.x)
        dst[idx] = row[idx & 255];
}

__global__ void zero_f32(float* p, long long n) {
    for (long long i = (long long)blockIdx.x * blockDim.x + threadIdx.x;
         i < n; i += (long long)gridDim.x * blockDim.x) p[i] = 0.f;
}
__global__ void zero_f16(half_t* p, long long n) {
    for (long long i = (long long)blockIdx.x * blockDim.x + threadIdx.x;
         i < n; i += (long long)gridDim.x * blockDim.x) p[i] = (half_t)0.f;
}

// ge[r] = G1[r] . ga2_w + b
__global__ void ga2_kernel(const float* __restrict__ G1,
                           const float* __restrict__ w,
                           const float* __restrict__ bsc,
                           float* __restrict__ ge) {
    int r = blockIdx.x * blockDim.x + threadIdx.x;
    if (r >= 8192) return;
    const float* row = G1 + (size_t)r * 256;
    float acc = bsc[0];
    for (int d = 0; d < 256; ++d) acc += row[d] * w[d];
    ge[r] = acc;
}

// masked softmax over S, pooled hidden, out linear [B,2]
__global__ void final_pool(const float* __restrict__ ge,
                           const int* __restrict__ lengths,
                           const float* __restrict__ hiddens,
                           const float* __restrict__ outw,
                           const float* __restrict__ outb,
                           float* __restrict__ out) {
    __shared__ float a[64];
    __shared__ float red[256];
    int b = blockIdx.x, t = threadIdx.x;
    if (t < 64) a[t] = (t < lengths[b]) ? ge[(size_t)b * 64 + t] : -3.4e38f;
    __syncthreads();
    if (t == 0) {
        float mx = -3.4e38f;
        for (int j = 0; j < 64; ++j) mx = fmaxf(mx, a[j]);
        float sum = 0.f;
        for (int j = 0; j < 64; ++j) { float ex = __expf(a[j] - mx); a[j] = ex; sum += ex; }
        float inv = 1.f / sum;
        for (int j = 0; j < 64; ++j) a[j] *= inv;
    }
    __syncthreads();
    float pooled = 0.f;
    for (int j = 0; j < 64; ++j)
        pooled += a[j] * hiddens[((size_t)b * 64 + j) * 256 + t];
    red[t] = pooled * outw[t * 2 + 0]; __syncthreads();
    for (int off = 128; off > 0; off >>= 1) {
        if (t < off) red[t] += red[t + off];
        __syncthreads();
    }
    float o0 = red[0] + outb[0];
    __syncthreads();
    red[t] = pooled * outw[t * 2 + 1]; __syncthreads();
    for (int off = 128; off > 0; off >>= 1) {
        if (t < off) red[t] += red[t + off];
        __syncthreads();
    }
    if (t == 0) {
        out[b * 2 + 0] = o0;
        out[b * 2 + 1] = red[0] + outb[1];
    }
}

// ---------------------------------------------------------------------------
// Host orchestration
// ---------------------------------------------------------------------------
enum {
    IN_SEQS = 0, IN_LEN, IN_TIME,
    P_EMB, P_TGT,
    P_TL_W, P_TL_B, P_TL2_W, P_TL2_B,
    P_CS1_W, P_CS1_B, P_CS2_W, P_CS2_B,
    P_VS1_W, P_VS1_B, P_VS2_W, P_VS2_B,
    P_CQ_W, P_CQ_B, P_FC_W, P_FC_B,
    P_MEM_W, P_MEM_B, P_STP_W, P_STP_B,
    P_FF1_W, P_FF1_B, P_FF2_W, P_FF2_B,
    P_GA1_W, P_GA1_B, P_GA2_W, P_GA2_B,
    P_OUT_W, P_OUT_B,
    P_LSTM_WIH, P_LSTM_WHH, P_GRU_WIH, P_GRU_WHH,
    P_MEM_G, P_MEM_BETA, P_STP_G, P_STP_BETA,
    P_LN_G, P_LN_BETA, P_LN2_G, P_LN2_BETA
};

extern "C" void kernel_launch(void* const* d_in, const int* in_sizes, int n_in,
                              void* d_out, int out_size, void* d_ws, size_t ws_size,
                              hipStream_t stream) {
    (void)in_sizes; (void)n_in; (void)out_size; (void)ws_size;
    const int B = 128, S = 64, D = 256, VP = 20001, VPAD = 20016;

    const int*   seqs    = (const int*)d_in[IN_SEQS];
    const int*   lengths = (const int*)d_in[IN_LEN];
    const float* emb     = (const float*)d_in[P_EMB];
    const float* tgt     = (const float*)d_in[P_TGT];
    const float* tl_w    = (const float*)d_in[P_TL_W];
    const float* tl_b    = (const float*)d_in[P_TL_B];
    const float* tl2_w   = (const float*)d_in[P_TL2_W];
    const float* tl2_b   = (const float*)d_in[P_TL2_B];
    const float* cs1_w   = (const float*)d_in[P_CS1_W];
    const float* cs1_b   = (const float*)d_in[P_CS1_B];
    const float* cs2_w   = (const float*)d_in[P_CS2_W];
    const float* cs2_b   = (const float*)d_in[P_CS2_B];
    const float* vs1_w   = (const float*)d_in[P_VS1_W];
    const float* vs1_b   = (const float*)d_in[P_VS1_B];
    const float* vs2_w   = (const float*)d_in[P_VS2_W];
    const float* vs2_b   = (const float*)d_in[P_VS2_B];
    const float* cq_w    = (const float*)d_in[P_CQ_W];
    const float* cq_b    = (const float*)d_in[P_CQ_B];
    const float* fc_w    = (const float*)d_in[P_FC_W];
    const float* fc_b    = (const float*)d_in[P_FC_B];
    const float* mem_w   = (const float*)d_in[P_MEM_W];
    const float* mem_b   = (const float*)d_in[P_MEM_B];
    const float* stp_w   = (const float*)d_in[P_STP_W];
    const float* stp_b   = (const float*)d_in[P_STP_B];
    const float* ff1_w   = (const float*)d_in[P_FF1_W];
    const float* ff1_b   = (const float*)d_in[P_FF1_B];
    const float* ff2_w   = (const float*)d_in[P_FF2_W];
    const float* ff2_b   = (const float*)d_in[P_FF2_B];
    const float* ga1_w   = (const float*)d_in[P_GA1_W];
    const float* ga1_b   = (const float*)d_in[P_GA1_B];
    const float* ga2_w   = (const float*)d_in[P_GA2_W];
    const float* ga2_b   = (const float*)d_in[P_GA2_B];
    const float* out_w   = (const float*)d_in[P_OUT_W];
    const float* out_b   = (const float*)d_in[P_OUT_B];
    const float* lstm_wih = (const float*)d_in[P_LSTM_WIH];
    const float* lstm_whh = (const float*)d_in[P_LSTM_WHH];
    const float* gru_wih  = (const float*)d_in[P_GRU_WIH];
    const float* gru_whh  = (const float*)d_in[P_GRU_WHH];
    const float* mem_g    = (const float*)d_in[P_MEM_G];
    const float* mem_beta = (const float*)d_in[P_MEM_BETA];
    const float* stp_g    = (const float*)d_in[P_STP_G];
    const float* stp_beta = (const float*)d_in[P_STP_BETA];
    const float* ln_g     = (const float*)d_in[P_LN_G];
    const float* ln_beta  = (const float*)d_in[P_LN_BETA];
    const float* ln2_g    = (const float*)d_in[P_LN2_G];
    const float* ln2_beta = (const float*)d_in[P_LN2_BETA];

    // ---- workspace bump allocator ----
    char* base = (char*)d_ws;
    size_t off = 0;
    auto alloc = [&](size_t bytes) -> void* {
        void* p = base + off;
        off += (bytes + 255) & ~(size_t)255;
        return p;
    };

    half_t* Wt_cs1 = (half_t*)alloc((size_t)512 * 256 * 2);
    half_t* Wt_vs1 = (half_t*)alloc((size_t)800 * 256 * 2);
    half_t* Wt_cq  = (half_t*)alloc((size_t)544 * 256 * 2);
    half_t* Wt_fc  = (half_t*)alloc((size_t)256 * 256 * 2);
    half_t* Wt_wih = (half_t*)alloc((size_t)256 * 768 * 2);
    half_t* Wt_whh = (half_t*)alloc((size_t)256 * 768 * 2);
    half_t* Wt_stp = (half_t*)alloc((size_t)256 * 256 * 2);
    half_t* Wt_mem = (half_t*)alloc((size_t)256 * 256 * 2);
    half_t* Wt_ff1 = (half_t*)alloc((size_t)256 * 1024 * 2);
    half_t* Wt_ff2 = (half_t*)alloc((size_t)1024 * 256 * 2);
    half_t* Wt_ga1 = (half_t*)alloc((size_t)512 * 256 * 2);

    half_t* A_cs   = (half_t*)alloc((size_t)VPAD * 512 * 2);
    float*  T_cs   = (float*)alloc((size_t)VPAD * 256 * 4);
    float*  pmaskV = (float*)alloc((size_t)VPAD * 4);
    float*  selected  = (float*)alloc((size_t)B * S * D * 4);
    float*  following = (float*)alloc((size_t)B * S * 32 * 4);
    float*  tgt1  = (float*)alloc(256 * 4);
    float*  tgt2  = (float*)alloc(256 * 4);
    float*  hx    = (float*)alloc((size_t)B * D * 4);
    half_t* hx16  = (half_t*)alloc((size_t)B * D * 2);
    float*  skipv = (float*)alloc(B * 4);
    half_t* A_vs  = (half_t*)alloc((size_t)B * 800 * 2);
    half_t* A_cq  = (half_t*)alloc((size_t)B * 544 * 2);
    float*  Tvs   = (float*)alloc((size_t)B * 256 * 4);
    float*  q     = (float*)alloc((size_t)B * 256 * 4);
    half_t* Az    = (half_t*)alloc((size_t)B * 256 * 2);
    float*  z     = (float*)alloc((size_t)B * 256 * 4);
    half_t* z16   = (half_t*)alloc((size_t)B * 256 * 2);
    float*  GI    = (float*)alloc((size_t)B * 768 * 4);
    float*  GH    = (float*)alloc((size_t)B * 768 * 4);
    float*  shx   = (float*)alloc((size_t)B * 256 * 4);
    half_t* shx16 = (half_t*)alloc((size_t)B * 256 * 2);
    float*  SHpre = (float*)alloc((size_t)B * 256 * 4);
    float*  sh    = (float*)alloc((size_t)B * 256 * 4);
    float*  h2    = (float*)alloc((size_t)B * 256 * 4);
    half_t* h2_16 = (half_t*)alloc((size_t)B * 256 * 2);
    float*  F1    = (float*)alloc((size_t)B * 1024 * 4);
    half_t* F1_16 = (half_t*)alloc((size_t)B * 1024 * 2);
    float*  F2    = (float*)alloc((size_t)B * 256 * 4);
    float*  hiddens = (float*)alloc((size_t)B * S * D * 4);
    float*  memory  = (float*)alloc((size_t)B * S * D * 4);
    float*  memrow  = (float*)alloc(256 * 4);
    float*  Mpre    = (float*)alloc((size_t)B * 256 * 4);
    half_t* A_ga    = (half_t*)alloc((size_t)B * S * 512 * 2);
    float*  G1f     = (float*)alloc((size_t)B * S * 256 * 4);
    float*  ge      = (float*)alloc((size_t)B * S * 4);

    auto gemm = [&](const half_t* A, const half_t* Wt, const float* bias,
                    float* C, half_t* C16, int M, int K, int N, int act) {
        dim3 g((N + 63) / 64, (M + 15) / 16);
        gemm_f16_wmma<<<g, dim3(128), 0, stream>>>(A, Wt, bias, C, C16, M, K, N, act);
    };
    auto tr = [&](const float* W, half_t* Wt, int K, int N) {
        long long total = (long long)K * N;
        int blocks = (int)((total + 255) / 256);
        if (blocks > 4096) blocks = 4096;
        transpose_f16<<<blocks, 256, 0, stream>>>(W, Wt, K, N);
    };
    auto pack = [&](half_t* A, int M, int K, int Mvalid,
                    const float* s0, int w0, int r0,
                    const float* s1, int w1, int r1,
                    const float* s2, int w2, int r2,
                    const float* s3, int w3, int r3) {
        long long total = (long long)M * K;
        int blocks = (int)((total + 255) / 256);
        if (blocks > 4096) blocks = 4096;
        pack_concat<<<blocks, 256, 0, stream>>>(A, total, K, Mvalid,
                                                s0, w0, r0, s1, w1, r1,
                                                s2, w2, r2, s3, w3, r3);
    };

    // ---- weight packing ----
    tr(cs1_w, Wt_cs1, 512, 256);
    tr(vs1_w, Wt_vs1, 800, 256);
    tr(cq_w,  Wt_cq,  544, 256);
    tr(fc_w,  Wt_fc,  256, 256);
    tr(gru_wih, Wt_wih, 256, 768);
    tr(gru_whh, Wt_whh, 256, 768);
    tr(stp_w, Wt_stp, 256, 256);
    tr(mem_w, Wt_mem, 256, 256);
    tr(ff1_w, Wt_ff1, 256, 1024);
    tr(ff2_w, Wt_ff2, 1024, 256);
    tr(ga1_w, Wt_ga1, 512, 256);

    // ---- vocab code-selection gate (big WMMA GEMM) ----
    pack(A_cs, VPAD, 512, VP, emb, 256, 256, tgt, 256, 0,
         nullptr, 0, 0, nullptr, 0, 0);
    gemm(A_cs, Wt_cs1, cs1_b, T_cs, nullptr, VPAD, 512, 256, /*tanh*/2);
    code_gate2<<<(VP + 255) / 256, 256, 0, stream>>>(T_cs, cs2_w, cs2_b, pmaskV, VP);

    // ---- selected, reverse LSTM, target projections ----
    build_selected<<<B * S, 256, 0, stream>>>(seqs, emb, pmaskV, selected);
    lstm_rev<<<B, 128, 0, stream>>>(selected, lstm_wih, lstm_whh, following);
    tl_kernel<<<1, 256, 0, stream>>>(tgt, tl_w, tl_b, tl2_w, tl2_b, tgt1, tgt2);

    // ---- initial state: memory = mem_ln(0) broadcast; hx = 0 ----
    memrow_kernel<<<1, 256, 0, stream>>>(mem_b, mem_g, mem_beta, memrow);
    fill_broadcast<<<2048, 256, 0, stream>>>(memrow, memory, (long long)B * S * D);
    zero_f32<<<128, 256, 0, stream>>>(hx, (long long)B * D);
    zero_f16<<<128, 256, 0, stream>>>(hx16, (long long)B * D);

    // ---- sequential steps ----
    for (int i = 0; i < S; ++i) {
        pack(A_vs, B, 800, B,
             hx, 256, 256,
             selected + (size_t)i * 256, 256, S * 256,
             following + (size_t)i * 32, 32, S * 32,
             tgt2, 256, 0);
        pack(A_cq, B, 544, B,
             hx, 256, 256,
             following + (size_t)i * 32, 32, S * 32,
             tgt1, 256, 0,
             nullptr, 0, 0);
        gemm(A_vs, Wt_vs1, vs1_b, Tvs, nullptr, B, 800, 256, 2);
        gemm(A_cq, Wt_cq, cq_b, q, nullptr, B, 544, 256, 1);
        vskip_kernel<<<1, 128, 0, stream>>>(Tvs, vs2_w, vs2_b, skipv);
        attn_fc_in<<<B, 256, 0, stream>>>(seqs, emb, pmaskV, q, i, Az);
        gemm(Az, Wt_fc, fc_b, z, z16, B, 256, 256, 0);
        gemm(z16, Wt_wih, nullptr, GI, nullptr, B, 256, 768, 0);
        gemm(hx16, Wt_whh, nullptr, GH, nullptr, B, 256, 768, 0);
        gru_combine<<<(B * 256) / 256, 256, 0, stream>>>(GI, GH, hx, skipv, shx, shx16);
        gemm(shx16, Wt_stp, stp_b, SHpre, nullptr, B, 256, 256, 1);
        layernorm256<<<B, 256, 0, stream>>>(SHpre, nullptr, stp_g, stp_beta,
                                            sh, nullptr, 256);
        mem_attention<<<B, 256, 0, stream>>>(sh, memory, ln2_g, ln2_beta, i, h2, h2_16);
        gemm(h2_16, Wt_ff1, ff1_b, F1, F1_16, B, 256, 1024, 1);
        gemm(F1_16, Wt_ff2, ff2_b, F2, nullptr, B, 1024, 256, 0);
        step_final<<<B, 256, 0, stream>>>(F2, h2, ln_g, ln_beta, shx, i,
                                          hx, hx16, hiddens);
        // incremental memory update: memory[:, i] = mem_ln(hx_new)
        gemm(hx16, Wt_mem, mem_b, Mpre, nullptr, B, 256, 256, 1);
        layernorm256<<<B, 256, 0, stream>>>(Mpre, nullptr, mem_g, mem_beta,
                                            memory + (size_t)i * 256, nullptr,
                                            (long long)S * 256);
    }

    // ---- final gated pooling ----
    pack(A_ga, B * S, 512, B * S, hiddens, 256, 256, tgt, 256, 0,
         nullptr, 0, 0, nullptr, 0, 0);
    gemm(A_ga, Wt_ga1, ga1_b, G1f, nullptr, B * S, 512, 256, 2);
    ga2_kernel<<<(B * S + 255) / 256, 256, 0, stream>>>(G1f, ga2_w, ga2_b, ge);
    final_pool<<<B, 256, 0, stream>>>(ge, lengths, hiddens, out_w, out_b,
                                      (float*)d_out);
}
